// ProgramLayer_27676769255907
// MI455X (gfx1250) — matrix operand order repaired and checked by reading, not compiled, not run on hardware
//
#include <hip/hip_runtime.h>
#include <hip/hip_bf16.h>

typedef __attribute__((ext_vector_type(16))) __bf16 v16bf;
typedef __attribute__((ext_vector_type(8)))  float  v8f;

#define D_DIM 1024
#define PD    32
#define NPAT  256
#define NTOK  2048
#define TOPK  4

static __device__ __forceinline__ v8f wmma_bf16(v16bf a, v16bf b, v8f c) {
  // D = A(16x32 bf16) * B(32x16 bf16) + C(16x16 f32)
  return __builtin_amdgcn_wmma_f32_16x16x32_bf16(false, a, false, b, (short)0, c,
                                                 false, false);
}

// Load a 16-element bf16 fragment from two 16-byte chunks (A layout).
static __device__ __forceinline__ v16bf load_frag2(const __bf16* c0,
                                                   const __bf16* c1) {
  union { v16bf v; uint4 q[2]; } u;
  u.q[0] = *(const uint4*)c0;
  u.q[1] = *(const uint4*)c1;
  return u.v;
}
// Load a 16-element bf16 fragment from one contiguous 32-byte chunk (B layout).
static __device__ __forceinline__ v16bf load_frag(const __bf16* c) {
  union { v16bf v; uint4 q[2]; } u;
  u.q[0] = ((const uint4*)c)[0];
  u.q[1] = ((const uint4*)c)[1];
  return u.v;
}

// ---------------------------------------------------------------------------
// One-time precision/layout conversion
// ---------------------------------------------------------------------------
__global__ void cvt_bf16_kernel(const float* __restrict__ in,
                                __bf16* __restrict__ out, int n) {
  int i = blockIdx.x * blockDim.x + threadIdx.x;
  if (i < n) out[i] = (__bf16)in[i];
}

// packed_vd[p][kstep(32)][ntile(2)][lane(32)][j(16)] : WMMA B-register order
__global__ void pack_vd_kernel(const float* __restrict__ vd,
                               __bf16* __restrict__ out) {
  int gid = blockIdx.x * blockDim.x + threadIdx.x;  // 256*32*2*32 threads
  const int lane  = gid & 31;
  const int ntile = (gid >> 5) & 1;
  const int kstep = (gid >> 6) & 31;
  const int p     = gid >> 11;
  const int m     = lane & 15;
  const int khalf = lane >> 4;
  const int k0    = kstep * 32;
  __bf16* o = out + (size_t)gid * 16;
  for (int j = 0; j < 16; ++j)
    o[j] = (__bf16)vd[((size_t)p * D_DIM + k0 + 16 * khalf + j) * PD +
                      16 * ntile + m];
}

// packed_vu[p][nt(64)][lane(32)][j(16)]
__global__ void pack_vu_kernel(const float* __restrict__ vu,
                               __bf16* __restrict__ out) {
  int gid = blockIdx.x * blockDim.x + threadIdx.x;  // 256*64*32 threads
  const int lane  = gid & 31;
  const int nt    = (gid >> 5) & 63;
  const int p     = gid >> 11;
  const int m     = lane & 15;
  const int khalf = lane >> 4;
  __bf16* o = out + (size_t)gid * 16;
  for (int j = 0; j < 16; ++j)
    o[j] = (__bf16)vu[(size_t)p * PD * D_DIM + (size_t)(16 * khalf + j) * D_DIM +
                      nt * 16 + m];
}

// ---------------------------------------------------------------------------
// Routing: h = x*W^T (WMMA K=1024), sim = h*keys^T (WMMA), top-4 + softmax.
// One wave per 16 tokens; all operands pre-converted bf16.
// ---------------------------------------------------------------------------
__global__ __launch_bounds__(32) void routing_kernel(
    const __bf16* __restrict__ xb, const __bf16* __restrict__ hwb,
    const __bf16* __restrict__ keysb, int* __restrict__ idx_out,
    float* __restrict__ w_out) {
  __shared__ float h_lds[16][PD];
  __shared__ float sim_lds[16][NPAT];
  const int lane  = threadIdx.x;
  const int m     = lane & 15;
  const int khalf = lane >> 4;
  const int t0    = blockIdx.x * 16;

  v8f acc0 = {}, acc1 = {};
  const __bf16* xr = xb + (size_t)(t0 + m) * D_DIM;
  for (int k0 = 0; k0 < D_DIM; k0 += 32) {
    v16bf a  = load_frag2(xr + k0 + 16 * khalf, xr + k0 + 16 * khalf + 8);
    v16bf b0 = load_frag(hwb + (size_t)m * D_DIM + k0 + 16 * khalf);
    v16bf b1 = load_frag(hwb + (size_t)(16 + m) * D_DIM + k0 + 16 * khalf);
    acc0 = wmma_bf16(a, b0, acc0);
    acc1 = wmma_bf16(a, b1, acc1);
  }
  for (int r = 0; r < 8; ++r) {
    h_lds[r + 8 * khalf][m]      = acc0[r];
    h_lds[r + 8 * khalf][16 + m] = acc1[r];
  }
  __syncthreads();

  v16bf ah;
  for (int j = 0; j < 8; ++j) ah[j]     = (__bf16)h_lds[m][16 * khalf + j];
  for (int j = 0; j < 8; ++j) ah[8 + j] = (__bf16)h_lds[m][16 * khalf + 8 + j];
  for (int n1 = 0; n1 < 16; ++n1) {
    v16bf bk = load_frag(keysb + (size_t)(n1 * 16 + m) * PD + 16 * khalf);
    v8f s = {};
    s = wmma_bf16(ah, bk, s);
    for (int r = 0; r < 8; ++r)
      sim_lds[r + 8 * khalf][n1 * 16 + m] = s[r];
  }
  __syncthreads();

  if (lane < 16) {
    const int tok = t0 + lane;
    float bv[TOPK]; int bi[TOPK];
    for (int k = 0; k < TOPK; ++k) { bv[k] = -3.4e38f; bi[k] = 0; }
    for (int p = 0; p < NPAT; ++p) {
      float v = sim_lds[lane][p];
      if (v > bv[TOPK - 1]) {
        int pos = TOPK - 1;
        while (pos > 0 && bv[pos - 1] < v) {
          bv[pos] = bv[pos - 1]; bi[pos] = bi[pos - 1]; --pos;
        }
        bv[pos] = v; bi[pos] = p;
      }
    }
    float e[TOPK], s = 0.f;
    for (int k = 0; k < TOPK; ++k) { e[k] = __expf(bv[k] - bv[0]); s += e[k]; }
    float inv = 1.f / s;
    for (int k = 0; k < TOPK; ++k) {
      idx_out[tok * TOPK + k] = bi[k];
      w_out[tok * TOPK + k]   = e[k] * inv;
    }
  }
}

// ---------------------------------------------------------------------------
// Grouped-GEMM bookkeeping
// ---------------------------------------------------------------------------
__global__ void zero_kernel(float* p, int n) {
  int i = blockIdx.x * blockDim.x + threadIdx.x;
  if (i < n) p[i] = 0.f;
}

__global__ void count_kernel(const int* __restrict__ idx, int* __restrict__ counts) {
  int a = blockIdx.x * blockDim.x + threadIdx.x;
  if (a < NTOK * TOPK) atomicAdd(&counts[idx[a]], 1);
}

__global__ void scan_kernel(const int* __restrict__ counts, int* __restrict__ offsets) {
  if (threadIdx.x == 0 && blockIdx.x == 0) {
    int run = 0;
    for (int p = 0; p < NPAT; ++p) { offsets[p] = run; run += counts[p]; }
  }
}

__global__ void fill_kernel(const int* __restrict__ idx, const float* __restrict__ w,
                            const int* __restrict__ offsets, int* __restrict__ cursor,
                            int* __restrict__ tok_of_slot,
                            float* __restrict__ wgt_of_slot) {
  int a = blockIdx.x * blockDim.x + threadIdx.x;
  if (a < NTOK * TOPK) {
    int p = idx[a];
    int pos = atomicAdd(&cursor[p], 1);
    int slot = offsets[p] + pos;
    tok_of_slot[slot] = a / TOPK;
    wgt_of_slot[slot] = w[a];
  }
}

// ---------------------------------------------------------------------------
// Grouped expert GEMM: block per pattern, 4 waves. All operands are pre-packed
// bf16 in exact WMMA register order -> hot loop is b128 loads + v_wmma only.
// ---------------------------------------------------------------------------
__global__ __launch_bounds__(128) void expert_kernel(
    const __bf16* __restrict__ xb, const __bf16* __restrict__ pvd,
    const __bf16* __restrict__ pvu, const int* __restrict__ counts,
    const int* __restrict__ offsets, const int* __restrict__ tok_of_slot,
    const float* __restrict__ wgt_of_slot, float* __restrict__ acc_out) {
  __shared__ float proj[16][PD];
  __shared__ int   tok_tile[16];
  __shared__ float wgt_tile[16];
  const int p     = blockIdx.x;
  const int cnt   = counts[p];
  const int base  = offsets[p];
  const int tid   = threadIdx.x;
  const int wave  = tid >> 5;
  const int lane  = tid & 31;
  const int m     = lane & 15;
  const int khalf = lane >> 4;

  for (int ts = 0; ts < cnt; ts += 16) {
    if (tid < 16) {
      int s = ts + tid;
      tok_tile[tid] = (s < cnt) ? tok_of_slot[base + s] : 0;
      wgt_tile[tid] = (s < cnt) ? wgt_of_slot[base + s] : 0.f;
    }
    for (int i = tid; i < 16 * PD; i += 128) ((float*)proj)[i] = 0.f;
    __syncthreads();

    // ---- down: proj = x_tile * values_down[p] ; wave owns 256 of K=1024 ----
    v8f d0 = {}, d1 = {};
    const __bf16* xr = xb + (size_t)tok_tile[m] * D_DIM;
    for (int kk = 0; kk < 256; kk += 32) {
      const int k0    = wave * 256 + kk;
      const int kstep = k0 >> 5;
      v16bf a = load_frag2(xr + k0 + 16 * khalf, xr + k0 + 16 * khalf + 8);
      const __bf16* bb =
          pvd + ((((size_t)p * 32 + kstep) * 2) * 32 + lane) * 16;
      v16bf b0 = load_frag(bb);
      v16bf b1 = load_frag(bb + 32 * 16);  // ntile 1
      d0 = wmma_bf16(a, b0, d0);
      d1 = wmma_bf16(a, b1, d1);
    }
    for (int r = 0; r < 8; ++r) {
      atomicAdd(&proj[r + 8 * khalf][m],      d0[r]);
      atomicAdd(&proj[r + 8 * khalf][16 + m], d1[r]);
    }
    __syncthreads();

    // ---- up: out = silu(proj) * values_up[p] ; wave owns 16 of 64 N-tiles --
    v16bf ap;
    for (int j = 0; j < 8; ++j) {
      float v0 = proj[m][16 * khalf + j];
      float v1 = proj[m][16 * khalf + 8 + j];
      ap[j]     = (__bf16)(v0 / (1.f + __expf(-v0)));
      ap[8 + j] = (__bf16)(v1 / (1.f + __expf(-v1)));
    }
    for (int nt = wave * 16; nt < wave * 16 + 16; ++nt) {
      const int ncol = nt * 16 + m;
      v16bf b = load_frag(pvu + (((size_t)p * 64 + nt) * 32 + lane) * 16);
      v8f o = {};
      o = wmma_bf16(ap, b, o);
      for (int r = 0; r < 8; ++r) {
        const int mm = r + 8 * khalf;
        if (ts + mm < cnt)
          atomicAdd(&acc_out[(size_t)tok_tile[mm] * D_DIM + ncol],
                    wgt_tile[mm] * o[r]);
      }
    }
    __syncthreads();
  }
}

__global__ void final_kernel(const float* __restrict__ x,
                             const float* __restrict__ acc,
                             const float* __restrict__ scale,
                             float* __restrict__ out) {
  int i = blockIdx.x * blockDim.x + threadIdx.x;
  if (i < NTOK * D_DIM) out[i] = x[i] + scale[0] * acc[i];
}

// ---------------------------------------------------------------------------
extern "C" void kernel_launch(void* const* d_in, const int* in_sizes, int n_in,
                              void* d_out, int out_size, void* d_ws, size_t ws_size,
                              hipStream_t stream) {
  (void)in_sizes; (void)n_in; (void)out_size; (void)ws_size;
  const float* x     = (const float*)d_in[0];
  const float* hw    = (const float*)d_in[1];
  const float* keys  = (const float*)d_in[2];
  const float* vdown = (const float*)d_in[3];
  const float* vup   = (const float*)d_in[4];
  const float* scale = (const float*)d_in[5];
  float* out = (float*)d_out;

  char* ws = (char*)d_ws;
  int*    idx_topk    = (int*)(ws);                          // 8192 int
  float*  w_topk      = (float*)(ws + 32768);                // 8192 f32
  int*    counts      = (int*)(ws + 65536);                  // 256 int
  int*    offsets     = (int*)(ws + 65536 + 1024);           // 256 int
  int*    cursor      = (int*)(ws + 65536 + 2048);           // 256 int
  int*    tok_of_slot = (int*)(ws + 65536 + 4096);           // 8192 int
  float*  wgt_of_slot = (float*)(ws + 65536 + 4096 + 32768); // 8192 f32
  float*  acc   = (float*)(ws + (1ull << 20));               //  8 MB @ 1 MB
  __bf16* xbf   = (__bf16*)(ws + 10ull * 1048576);           //  4 MB @ 10 MB
  __bf16* hwbf  = (__bf16*)(ws + 14ull * 1048576);           // 64 KB @ 14 MB
  __bf16* keybf = (__bf16*)(ws + 14ull * 1048576 + 65536);   // 16 KB
  __bf16* pvd   = (__bf16*)(ws + 15ull * 1048576);           // 16 MB @ 15 MB
  __bf16* pvu   = (__bf16*)(ws + 31ull * 1048576);           // 16 MB @ 31 MB

  zero_kernel<<<(NTOK * D_DIM + 255) / 256, 256, 0, stream>>>(acc, NTOK * D_DIM);
  zero_kernel<<<3, 256, 0, stream>>>((float*)counts, 768);  // counts+offsets+cursor

  cvt_bf16_kernel<<<(NTOK * D_DIM + 255) / 256, 256, 0, stream>>>(x, xbf,
                                                                  NTOK * D_DIM);
  cvt_bf16_kernel<<<(PD * D_DIM + 255) / 256, 256, 0, stream>>>(hw, hwbf,
                                                                PD * D_DIM);
  cvt_bf16_kernel<<<(NPAT * PD + 255) / 256, 256, 0, stream>>>(keys, keybf,
                                                               NPAT * PD);
  pack_vd_kernel<<<(NPAT * 32 * 2 * 32) / 256, 256, 0, stream>>>(vdown, pvd);
  pack_vu_kernel<<<(NPAT * 64 * 32) / 256, 256, 0, stream>>>(vup, pvu);

  routing_kernel<<<NTOK / 16, 32, 0, stream>>>(xbf, hwbf, keybf, idx_topk,
                                               w_topk);
  count_kernel<<<(NTOK * TOPK + 255) / 256, 256, 0, stream>>>(idx_topk, counts);
  scan_kernel<<<1, 32, 0, stream>>>(counts, offsets);
  fill_kernel<<<(NTOK * TOPK + 255) / 256, 256, 0, stream>>>(
      idx_topk, w_topk, offsets, cursor, tok_of_slot, wgt_of_slot);
  expert_kernel<<<NPAT, 128, 0, stream>>>(xbf, pvd, pvu, counts, offsets,
                                          tok_of_slot, wgt_of_slot, acc);
  final_kernel<<<(NTOK * D_DIM + 255) / 256, 256, 0, stream>>>(x, acc, scale, out);
}
